// DisentangledVQ_24739011625046
// MI455X (gfx1250) — compile-verified, matrix operand order checked by reading, not run on hardware
//
#include <hip/hip_runtime.h>
#include <hip/hip_bf16.h>
#include <math.h>

typedef __attribute__((ext_vector_type(16))) _Float16 v16h;
typedef __attribute__((ext_vector_type(8)))  float    v8f;

constexpr int KD       = 1024;   // d_model
constexpr int KH       = 512;    // half
constexpr int NROWS    = 16384;  // B*S = 8*2048
constexpr int NCONTENT = 4096;
constexpr int NSTYLE   = 1024;
constexpr float LN_EPS   = 1e-5f;
constexpr float NORM_EPS = 1e-12f;

// Load one lane's 16 f16 values of a 16x32 WMMA A/B tile from K-contiguous memory.
// ISA 7.12.2 layout: half h=lane>>4 holds K = {kb+8h .. +7} and {kb+16+8h .. +7},
// i.e. two contiguous 16-byte chunks -> two global_load_b128.
__device__ __forceinline__ v16h load16(const _Float16* __restrict__ p, int kb, int h) {
    union { uint4 q[2]; v16h v; } u;
    u.q[0] = *(const uint4*)(p + kb + 8 * h);
    u.q[1] = *(const uint4*)(p + kb + 16 + 8 * h);
    return u.v;
}

__device__ __forceinline__ v8f wmma16(v16h a, v16h b, v8f c) {
    return __builtin_amdgcn_wmma_f32_16x16x32_f16(false, a, false, b, (short)0, c, false, false);
}

// ---------------- kernel 0: LN stats + xhat = (x-mu)*rstd  (f16) ----------------
__global__ void k_ln_xhat(const float* __restrict__ x, _Float16* __restrict__ xhat) {
    int row = blockIdx.x, tid = threadIdx.x;
    __shared__ float sm[256];
    float s = 0.f;
    for (int i = tid; i < KD; i += 256) s += x[row * KD + i];
    sm[tid] = s; __syncthreads();
    for (int o = 128; o > 0; o >>= 1) { if (tid < o) sm[tid] += sm[tid + o]; __syncthreads(); }
    float m = sm[0] / KD;
    __syncthreads();
    float v = 0.f;
    for (int i = tid; i < KD; i += 256) { float d = x[row * KD + i] - m; v += d * d; }
    sm[tid] = v; __syncthreads();
    for (int o = 128; o > 0; o >>= 1) { if (tid < o) sm[tid] += sm[tid + o]; __syncthreads(); }
    float r = rsqrtf(sm[0] / KD + LN_EPS);
    for (int i = tid; i < KD; i += 256)
        xhat[row * KD + i] = (_Float16)((x[row * KD + i] - m) * r);
}

// ---------------- kernel 1: codebook L2-normalize -> f16 ----------------
__global__ void k_cb_norm(const float* __restrict__ cb, _Float16* __restrict__ out) {
    int code = blockIdx.x, tid = threadIdx.x;
    __shared__ float sm[256];
    float s = 0.f;
    for (int i = tid; i < KH; i += 256) { float v = cb[code * KH + i]; s += v * v; }
    sm[tid] = s; __syncthreads();
    for (int o = 128; o > 0; o >>= 1) { if (tid < o) sm[tid] += sm[tid + o]; __syncthreads(); }
    float inv = 1.f / fmaxf(sqrtf(sm[0]), NORM_EPS);
    __syncthreads();
    for (int i = tid; i < KH; i += 256) out[code * KH + i] = (_Float16)(cb[code * KH + i] * inv);
}

// ---------------- kernel 2: weight transpose -> f16, optionally folding LN gain ----------------
__global__ void k_wT(const float* __restrict__ w, const float* __restrict__ g,
                     _Float16* __restrict__ wT, int K, int N, int useG) {
    int n = blockIdx.x * 16 + threadIdx.x;
    int k = blockIdx.y * 16 + threadIdx.y;
    float v = w[(size_t)k * N + n];
    if (useG) v *= g[k];
    wT[(size_t)n * K + k] = (_Float16)v;
}

// ---------------- kernel 3: fold LN shift through weights into the bias ----------------
__global__ void k_bias_fold(const float* __restrict__ lnb, const float* __restrict__ w,
                            const float* __restrict__ bias, float* __restrict__ out, int K, int N) {
    int n = blockIdx.x * blockDim.x + threadIdx.x;
    if (n >= N) return;
    float s = bias[n];
    for (int k = 0; k < K; ++k) s += lnb[k] * w[k * N + n];
    out[n] = s;
}

// ---------------- kernel 4: projection GEMM 2x2-blocked: tanh(xhat @ wT^T + bias') ----------------
// block = 4 waves, each wave a 32x32 output tile (4 WMMAs / 8 loads per K-slab).
// grid = (NROWS/32, KH/128)
__global__ __launch_bounds__(128)
void k_proj(const _Float16* __restrict__ xhat, const _Float16* __restrict__ wT,
            const float* __restrict__ bias, _Float16* __restrict__ feat) {
    int wave = threadIdx.x >> 5, lane = threadIdx.x & 31;
    int m = lane & 15, h = lane >> 4;
    int rowBase = blockIdx.x * 32;
    int colBase = blockIdx.y * 128 + wave * 32;
    const _Float16* pa0 = xhat + (size_t)(rowBase + m) * KD;
    const _Float16* pa1 = pa0 + (size_t)16 * KD;
    const _Float16* pb0 = wT + (size_t)(colBase + m) * KD;
    const _Float16* pb1 = pb0 + (size_t)16 * KD;
    v8f acc00 = {}, acc01 = {}, acc10 = {}, acc11 = {};
#pragma unroll 2
    for (int kb = 0; kb < KD; kb += 32) {
        v16h a0 = load16(pa0, kb, h);
        v16h a1 = load16(pa1, kb, h);
        v16h b0 = load16(pb0, kb, h);
        v16h b1 = load16(pb1, kb, h);
        acc00 = wmma16(a0, b0, acc00);
        acc01 = wmma16(a0, b1, acc01);
        acc10 = wmma16(a1, b0, acc10);
        acc11 = wmma16(a1, b1, acc11);
    }
    int n = lane & 15;
#pragma unroll
    for (int r = 0; r < 8; ++r) {
        int r0 = rowBase + r + 8 * h, r1 = r0 + 16;
        int c0 = colBase + n,         c1 = c0 + 16;
        feat[(size_t)r0 * KH + c0] = (_Float16)tanhf(acc00[r] + bias[c0]);
        feat[(size_t)r0 * KH + c1] = (_Float16)tanhf(acc01[r] + bias[c1]);
        feat[(size_t)r1 * KH + c0] = (_Float16)tanhf(acc10[r] + bias[c0]);
        feat[(size_t)r1 * KH + c1] = (_Float16)tanhf(acc11[r] + bias[c1]);
    }
}

// ---------------- kernel 5: per-row feature stats + disentangle accum ----------------
__global__ void k_feat_stats(const _Float16* __restrict__ fc, const _Float16* __restrict__ fs,
                             float* __restrict__ acc) {
    int row = blockIdx.x, tid = threadIdx.x;
    __shared__ float s0[256], s1[256], s2[256];
    float a = 0.f, b = 0.f, c = 0.f;
    for (int i = tid; i < KH; i += 256) {
        float vc = (float)fc[(size_t)row * KH + i];
        float vs = (float)fs[(size_t)row * KH + i];
        a += vc * vc; b += vs * vs; c += vc * vs;
    }
    s0[tid] = a; s1[tid] = b; s2[tid] = c; __syncthreads();
    for (int o = 128; o > 0; o >>= 1) {
        if (tid < o) { s0[tid] += s0[tid + o]; s1[tid] += s1[tid + o]; s2[tid] += s2[tid + o]; }
        __syncthreads();
    }
    if (tid == 0) {
        atomicAdd(&acc[0], s0[0]);
        atomicAdd(&acc[2], s1[0]);
        float cs = fabsf(s2[0]) / (fmaxf(sqrtf(s0[0]), NORM_EPS) * fmaxf(sqrtf(s1[0]), NORM_EPS));
        atomicAdd(&acc[4], cs);
    }
}

// ---------------- kernel 6: WMMA VQ search, 2 code-tiles per pass ----------------
// block = 8 waves over 16 rows; each wave scans 32-code chunks (A reused across both tiles).
__global__ __launch_bounds__(256)
void k_vq(const _Float16* __restrict__ feat, const _Float16* __restrict__ cbn, int K,
          int* __restrict__ best_idx, float* __restrict__ idx_out_f,
          float* __restrict__ acc, int slot) {
    int wave = threadIdx.x >> 5, lane = threadIdx.x & 31;
    int m = lane & 15, h = lane >> 4;
    int rowBase = blockIdx.x * 16;
    const _Float16* pa = feat + (size_t)(rowBase + m) * KH;
    float bestv[8]; int besti[8];
#pragma unroll
    for (int r = 0; r < 8; ++r) { bestv[r] = -3.0e38f; besti[r] = 0; }

    for (int c0 = wave * 32; c0 < K; c0 += 256) {
        const _Float16* pb0 = cbn + (size_t)(c0 + m) * KH;
        const _Float16* pb1 = pb0 + (size_t)16 * KH;
        v8f s0 = {}, s1 = {};
#pragma unroll 2
        for (int kb = 0; kb < KH; kb += 32) {
            v16h a  = load16(pa, kb, h);
            v16h b0 = load16(pb0, kb, h);
            v16h b1 = load16(pb1, kb, h);
            s0 = wmma16(a, b0, s0);
            s1 = wmma16(a, b1, s1);
        }
        int code0 = c0 + (lane & 15);
#pragma unroll
        for (int r = 0; r < 8; ++r) {
            float v0 = s0[r];
            if (v0 > bestv[r]) { bestv[r] = v0; besti[r] = code0; }
            float v1 = s1[r];
            if (v1 > bestv[r]) { bestv[r] = v1; besti[r] = code0 + 16; }
        }
    }
    // argmax across 16 lanes of each half (tie-break -> lowest index, like jnp.argmax)
#pragma unroll
    for (int r = 0; r < 8; ++r) {
#pragma unroll
        for (int off = 1; off < 16; off <<= 1) {
            float ov = __shfl_xor(bestv[r], off, 32);
            int   oi = __shfl_xor(besti[r], off, 32);
            if (ov > bestv[r] || (ov == bestv[r] && oi < besti[r])) { bestv[r] = ov; besti[r] = oi; }
        }
    }
    __shared__ float smV[8][16];
    __shared__ int   smI[8][16];
    if ((lane & 15) == 0) {
#pragma unroll
        for (int r = 0; r < 8; ++r) { smV[wave][r + 8 * h] = bestv[r]; smI[wave][r + 8 * h] = besti[r]; }
    }
    __syncthreads();
    if (threadIdx.x < 16) {
        int mm = threadIdx.x;
        float bv = smV[0][mm]; int bi = smI[0][mm];
        for (int wv = 1; wv < 8; ++wv) {
            float v = smV[wv][mm]; int i2 = smI[wv][mm];
            if (v > bv || (v == bv && i2 < bi)) { bv = v; bi = i2; }
        }
        int orow = rowBase + mm;
        best_idx[orow]  = bi;
        idx_out_f[orow] = (float)bi;
        atomicAdd(&acc[slot], bv);
    }
}

// ---------------- kernel 7: combined GEMM 2x2-blocked (gathered codewords) ----------------
// block = 8 waves, wave = 32x32 tile. grid = (NROWS/32, KD/256)
__global__ __launch_bounds__(256)
void k_comb(const _Float16* __restrict__ cbc, const _Float16* __restrict__ cbs,
            const int* __restrict__ idxc, const int* __restrict__ idxs,
            const _Float16* __restrict__ wT, const float* __restrict__ bias,
            float* __restrict__ outpre) {
    int wave = threadIdx.x >> 5, lane = threadIdx.x & 31;
    int m = lane & 15, h = lane >> 4;
    int rowBase = blockIdx.x * 32;
    int colBase = blockIdx.y * 256 + wave * 32;
    int row0 = rowBase + m, row1 = row0 + 16;
    const _Float16* pac0 = cbc + (size_t)idxc[row0] * KH;
    const _Float16* pas0 = cbs + (size_t)idxs[row0] * KH;
    const _Float16* pac1 = cbc + (size_t)idxc[row1] * KH;
    const _Float16* pas1 = cbs + (size_t)idxs[row1] * KH;
    const _Float16* pb0  = wT + (size_t)(colBase + m) * KD;
    const _Float16* pb1  = pb0 + (size_t)16 * KD;
    v8f acc00 = {}, acc01 = {}, acc10 = {}, acc11 = {};
#pragma unroll 2
    for (int kb = 0; kb < KD; kb += 32) {
        v16h a0 = (kb < KH) ? load16(pac0, kb, h) : load16(pas0, kb - KH, h);
        v16h a1 = (kb < KH) ? load16(pac1, kb, h) : load16(pas1, kb - KH, h);
        v16h b0 = load16(pb0, kb, h);
        v16h b1 = load16(pb1, kb, h);
        acc00 = wmma16(a0, b0, acc00);
        acc01 = wmma16(a0, b1, acc01);
        acc10 = wmma16(a1, b0, acc10);
        acc11 = wmma16(a1, b1, acc11);
    }
    int n = lane & 15;
#pragma unroll
    for (int r = 0; r < 8; ++r) {
        int r0 = rowBase + r + 8 * h, r1 = r0 + 16;
        int c0 = colBase + n,         c1 = c0 + 16;
        outpre[(size_t)r0 * KD + c0] = acc00[r] + bias[c0];
        outpre[(size_t)r0 * KD + c1] = acc01[r] + bias[c1];
        outpre[(size_t)r1 * KD + c0] = acc10[r] + bias[c0];
        outpre[(size_t)r1 * KD + c1] = acc11[r] + bias[c1];
    }
}

// ---------------- kernel 8: output LayerNorm ----------------
__global__ void k_out_ln(const float* __restrict__ pre, const float* __restrict__ g,
                         const float* __restrict__ b, float* __restrict__ out) {
    int row = blockIdx.x, tid = threadIdx.x;
    __shared__ float sm[256];
    float s = 0.f;
    for (int i = tid; i < KD; i += 256) s += pre[(size_t)row * KD + i];
    sm[tid] = s; __syncthreads();
    for (int o = 128; o > 0; o >>= 1) { if (tid < o) sm[tid] += sm[tid + o]; __syncthreads(); }
    float mean = sm[0] / KD;
    __syncthreads();
    float v = 0.f;
    for (int i = tid; i < KD; i += 256) { float d = pre[(size_t)row * KD + i] - mean; v += d * d; }
    sm[tid] = v; __syncthreads();
    for (int o = 128; o > 0; o >>= 1) { if (tid < o) sm[tid] += sm[tid + o]; __syncthreads(); }
    float r = rsqrtf(sm[0] / KD + LN_EPS);
    for (int i = tid; i < KD; i += 256)
        out[(size_t)row * KD + i] = (pre[(size_t)row * KD + i] - mean) * r * g[i] + b[i];
}

// ---------------- kernel 9: scalar finalize ----------------
__global__ void k_final(const float* __restrict__ acc,
                        float* __restrict__ out_loss, float* __restrict__ out_dis) {
    float elc  = ((float)NROWS - 2.f * acc[1] + acc[0]) / ((float)NROWS * (float)KH);
    float els  = ((float)NROWS - 2.f * acc[3] + acc[2]) / ((float)NROWS * (float)KH);
    float cosm = acc[4] / (float)NROWS;
    float dis  = fminf(fmaxf(cosm, 0.f), 1.f);
    *out_loss = 0.1f * elc + 0.1f * els + 0.5f * dis;
    *out_dis  = dis;
}

extern "C" void kernel_launch(void* const* d_in, const int* in_sizes, int n_in,
                              void* d_out, int out_size, void* d_ws, size_t ws_size,
                              hipStream_t stream) {
    (void)in_sizes; (void)n_in; (void)out_size; (void)ws_size;
    const float* x      = (const float*)d_in[0];
    const float* ln_c_g = (const float*)d_in[1];
    const float* ln_c_b = (const float*)d_in[2];
    const float* w_c    = (const float*)d_in[3];
    const float* b_c    = (const float*)d_in[4];
    const float* ln_s_g = (const float*)d_in[5];
    const float* ln_s_b = (const float*)d_in[6];
    const float* w_s    = (const float*)d_in[7];
    const float* b_s    = (const float*)d_in[8];
    const float* cb_c   = (const float*)d_in[9];
    const float* cb_s   = (const float*)d_in[10];
    const float* w_comb = (const float*)d_in[11];
    const float* b_comb = (const float*)d_in[12];
    const float* ln_o_g = (const float*)d_in[13];
    const float* ln_o_b = (const float*)d_in[14];

    char* ws = (char*)d_ws;
    size_t off = 0;
    auto alloc = [&](size_t bytes) -> void* {
        void* p = ws + off;
        off = (off + bytes + 255) & ~(size_t)255;
        return p;
    };
    float*    acc    = (float*)alloc(8 * sizeof(float));
    int*      idxc   = (int*)alloc(NROWS * sizeof(int));
    int*      idxs   = (int*)alloc(NROWS * sizeof(int));
    float*    biasC  = (float*)alloc(KH * sizeof(float));
    float*    biasS  = (float*)alloc(KH * sizeof(float));
    _Float16* xhat   = (_Float16*)alloc((size_t)NROWS * KD * sizeof(_Float16));
    _Float16* cbnc   = (_Float16*)alloc((size_t)NCONTENT * KH * sizeof(_Float16));
    _Float16* cbns   = (_Float16*)alloc((size_t)NSTYLE * KH * sizeof(_Float16));
    _Float16* wtC    = (_Float16*)alloc((size_t)KH * KD * sizeof(_Float16));
    _Float16* wtS    = (_Float16*)alloc((size_t)KH * KD * sizeof(_Float16));
    _Float16* wtComb = (_Float16*)alloc((size_t)KD * KD * sizeof(_Float16));
    _Float16* featc  = (_Float16*)alloc((size_t)NROWS * KH * sizeof(_Float16));
    _Float16* feats  = (_Float16*)alloc((size_t)NROWS * KH * sizeof(_Float16));
    float*    comb   = (float*)alloc((size_t)NROWS * KD * sizeof(float));

    float* out       = (float*)d_out;
    float* out_comb  = out;
    float* out_loss  = out + (size_t)NROWS * KD;
    float* out_idxc  = out_loss + 1;
    float* out_idxs  = out_idxc + NROWS;
    float* out_dis   = out_idxs + NROWS;

    hipMemsetAsync(acc, 0, 8 * sizeof(float), stream);

    k_ln_xhat<<<NROWS, 256, 0, stream>>>(x, xhat);
    k_cb_norm<<<NCONTENT, 256, 0, stream>>>(cb_c, cbnc);
    k_cb_norm<<<NSTYLE, 256, 0, stream>>>(cb_s, cbns);

    dim3 tb(16, 16);
    k_wT<<<dim3(KH / 16, KD / 16), tb, 0, stream>>>(w_c, ln_c_g, wtC, KD, KH, 1);
    k_wT<<<dim3(KH / 16, KD / 16), tb, 0, stream>>>(w_s, ln_s_g, wtS, KD, KH, 1);
    k_wT<<<dim3(KD / 16, KD / 16), tb, 0, stream>>>(w_comb, nullptr, wtComb, KD, KD, 0);
    k_bias_fold<<<(KH + 255) / 256, 256, 0, stream>>>(ln_c_b, w_c, b_c, biasC, KD, KH);
    k_bias_fold<<<(KH + 255) / 256, 256, 0, stream>>>(ln_s_b, w_s, b_s, biasS, KD, KH);

    k_proj<<<dim3(NROWS / 32, KH / 128), 128, 0, stream>>>(xhat, wtC, biasC, featc);
    k_proj<<<dim3(NROWS / 32, KH / 128), 128, 0, stream>>>(xhat, wtS, biasS, feats);

    k_feat_stats<<<NROWS, 256, 0, stream>>>(featc, feats, acc);

    k_vq<<<NROWS / 16, 256, 0, stream>>>(featc, cbnc, NCONTENT, idxc, out_idxc, acc, 1);
    k_vq<<<NROWS / 16, 256, 0, stream>>>(feats, cbns, NSTYLE, idxs, out_idxs, acc, 3);

    k_comb<<<dim3(NROWS / 32, KD / 256), 256, 0, stream>>>(cbnc, cbns, idxc, idxs, wtComb, b_comb, comb);
    k_out_ln<<<NROWS, 256, 0, stream>>>(comb, ln_o_g, ln_o_b, out_comb);
    k_final<<<1, 1, 0, stream>>>(acc, out_loss, out_dis);
}
